// VDEncoderDecoder_26414048871116
// MI455X (gfx1250) — matrix operand order repaired and checked by reading, not compiled, not run on hardware
//
#include <hip/hip_runtime.h>
#include <hip/hip_bf16.h>
#include <math.h>

typedef _Float16 h16;
typedef __attribute__((ext_vector_type(16))) _Float16 v16h;
typedef __attribute__((ext_vector_type(8)))  _Float16 v8h;
typedef __attribute__((ext_vector_type(8)))  float    v8f;

union V16 { v16h v; v8h half8[2]; h16 e[16]; };

// Branch-free fast activations: v_exp_f32 + v_rcp_f32, saturating correctly.
__device__ __forceinline__ float fast_sigm(float x) {
  return __builtin_amdgcn_rcpf(1.0f + __expf(-x));
}
__device__ __forceinline__ float fast_tanh(float x) {
  const float e = __expf(2.0f * x);
  return 1.0f - 2.0f * __builtin_amdgcn_rcpf(e + 1.0f);
}

#define T_ENC 96
#define T_DEC 128
#define BT    16   // batch tile per workgroup

struct __align__(16) SMem {
  union {                            // encoder phase vs FC phase
    h16   xh[BT][T_ENC][8];          // 24576 B : x tile in f16
    float fc1w[32][128];             // 16384 B : fc1 weights
  } u1;
  union {                            // encoder gates vs decoder gates
    float G[BT][256];                // 16384 B
    float Gd[BT][64];                //  4096 B
  } u2;
  h16   h1[BT][64];                  //  2048 B : encoder LSTM1 hidden (f16, A-ready)
  union {                            // partial sums vs fc2 weights
    float part[BT][8][4];            //  2048 B
    struct { float w[32][32]; float b[32]; } fc2; // 4224 B
  } u3;
  float h2s[BT], c2s[BT];            // encoder LSTM2 state (H=1)
  h16   dec_in[T_DEC][BT];           //  4096 B : relu(enc_out) ++ x_aux
  h16   hd[BT][16];                  //   512 B : decoder LSTM1 hidden
  float h2d[BT], c2d[BT];            // decoder LSTM2 state
  float dseq[T_DEC][BT];             //  8192 B : decoder output sequence
  float tmp[BT][32];                 //  2048 B : fc1 output
  float e2w[4][64]; float e2b[4]; float e2whh[4];
  float d2w[4][16]; float d2b[4]; float d2whh[4];
  alignas(16) h16 zpad[8];           // zero pad for branch-free A loads
};
static_assert(sizeof(SMem) <= 65536, "LDS budget");

__global__ __launch_bounds__(128) void lstm_encdec_fused(
    const float* __restrict__ x,
    const float* __restrict__ e1_Wih, const float* __restrict__ e1_Whh, const float* __restrict__ e1_b,
    const float* __restrict__ e2_Wih, const float* __restrict__ e2_Whh, const float* __restrict__ e2_b,
    const float* __restrict__ d1_Wih, const float* __restrict__ d1_Whh, const float* __restrict__ d1_b,
    const float* __restrict__ d2_Wih, const float* __restrict__ d2_Whh, const float* __restrict__ d2_b,
    const float* __restrict__ fc1_W,  const float* __restrict__ fc1_b,
    const float* __restrict__ fc2_W,  const float* __restrict__ fc2_b,
    float* __restrict__ out)
{
  __shared__ SMem s;
  const int tid  = threadIdx.x;
  const int wave = tid >> 5;
  const int lane = tid & 31;
  const int lh   = lane & 15;      // row (batch) / column-in-tile id
  const int hi   = lane >> 4;      // half-wave selector for WMMA K layout
  const int b0   = blockIdx.x * BT;

  // ---- stage in x tile (f32 -> f16) ----
  {
    const float* xp = x + (size_t)b0 * (T_ENC * 8);
    h16* dst = &s.u1.xh[0][0][0];
    for (int i = tid; i < BT * T_ENC * 8; i += 128) dst[i] = (h16)xp[i];
  }
  for (int i = tid; i < BT * 64; i += 128) (&s.h1[0][0])[i] = (h16)0.0f;
  if (tid < BT) { s.h2s[tid] = 0.f; s.c2s[tid] = 0.f; s.h2d[tid] = 0.f; s.c2d[tid] = 0.f; }
  for (int i = tid; i < 4 * 64; i += 128) (&s.e2w[0][0])[i] = e2_Wih[i];
  if (tid < 4) { s.e2b[tid] = e2_b[tid]; s.e2whh[tid] = e2_Whh[tid]; }
  for (int i = tid; i < 4 * 16; i += 128) (&s.d2w[0][0])[i] = d2_Wih[i];
  if (tid >= 4 && tid < 8)  s.d2b[tid - 4]  = d2_b[tid - 4];
  if (tid >= 8 && tid < 12) s.d2whh[tid - 8] = d2_Whh[tid - 8];
  if (tid < 8) s.zpad[tid] = (h16)0.0f;

  // ---- encoder LSTM1 B-matrices (f16) resident in registers ----
  // Augmented K = [ x(0..7) | pad(8..31) ] chunk0, then h(0..63) = chunks 1,2.
  // B 32x16 (this code's contiguous K order, A matches): lanes 0-15 hold K 0-15 of column lh, lanes 16-31 hold K 16-31.
  V16 bw[4][3];
  float bias[4];
  #pragma unroll
  for (int t4 = 0; t4 < 4; ++t4) {
    const int gate = wave * 64 + t4 * 16 + lh;   // wave w owns gate block w (i/f/g/o)
    bias[t4] = e1_b[gate];
    V16 c0;
    #pragma unroll
    for (int i = 0; i < 16; ++i) c0.e[i] = (h16)0.0f;
    if (hi == 0) {
      #pragma unroll
      for (int i = 0; i < 8; ++i) c0.e[i] = (h16)e1_Wih[gate * 8 + i];
    }
    bw[t4][0] = c0;
    V16 c1, c2;
    #pragma unroll
    for (int i = 0; i < 16; ++i) {
      c1.e[i] = (h16)e1_Whh[gate * 64 + hi * 16 + i];
      c2.e[i] = (h16)e1_Whh[gate * 64 + 32 + hi * 16 + i];
    }
    bw[t4][1] = c1;
    bw[t4][2] = c2;
  }

  v8h zero8;
  #pragma unroll
  for (int i = 0; i < 8; ++i) zero8[i] = (h16)0.0f;

  float cst[8];                       // encoder LSTM1 cell state, 8 units/thread
  #pragma unroll
  for (int q = 0; q < 8; ++q) cst[q] = 0.f;
  const int eb = tid >> 3;            // batch row for elementwise
  const int u0 = (tid & 7) * 8;       // unit base for elementwise
  const int p8 = tid & 7;             // partial slot

  __syncthreads();

  // ================= encoder time loop =================
  for (int t = 0; t < T_ENC; ++t) {
    // Build A operands (contiguous K order, matching B): lanes<16 hold K0-15; lanes>=16 K16-31
    V16 a0, a1, a2;
    const h16* xs = (hi == 0) ? &s.u1.xh[lh][t][0] : s.zpad;  // branch-free select
    a0.half8[0] = *(const v8h*)xs;
    a0.half8[1] = zero8;
    const v8h* ph = (const v8h*)&s.h1[lh][0];                 // 64 halves = 8 slots
    a1.half8[0] = ph[2 * hi];     a1.half8[1] = ph[2 * hi + 1];       // h dims 0-31
    a2.half8[0] = ph[4 + 2 * hi]; a2.half8[1] = ph[5 + 2 * hi];       // h dims 32-63

    #pragma unroll
    for (int t4 = 0; t4 < 4; ++t4) {
      v8f c;
      #pragma unroll
      for (int i = 0; i < 8; ++i) c[i] = bias[t4];
      c = __builtin_amdgcn_wmma_f32_16x16x32_f16(false, a0.v, false, bw[t4][0].v, (short)0, c, false, false);
      c = __builtin_amdgcn_wmma_f32_16x16x32_f16(false, a1.v, false, bw[t4][1].v, (short)0, c, false, false);
      c = __builtin_amdgcn_wmma_f32_16x16x32_f16(false, a2.v, false, bw[t4][2].v, (short)0, c, false, false);
      const int col = wave * 64 + t4 * 16 + lh;
      #pragma unroll
      for (int r = 0; r < 8; ++r) s.u2.G[r + hi * 8][col] = c[r];
    }
    __syncthreads();

    // LSTM1 elementwise fused with LSTM2 partial dots (h values stay in registers)
    {
      float q0 = 0.f, q1 = 0.f, q2 = 0.f, q3 = 0.f;
      #pragma unroll
      for (int q = 0; q < 8; ++q) {
        const int u = u0 + q;
        const float gi = s.u2.G[eb][u],       gf = s.u2.G[eb][64 + u];
        const float gg = s.u2.G[eb][128 + u], go = s.u2.G[eb][192 + u];
        const float cc = fast_sigm(gf) * cst[q] + fast_sigm(gi) * fast_tanh(gg);
        cst[q] = cc;
        const float hv = fast_sigm(go) * fast_tanh(cc);
        s.h1[eb][u] = (h16)hv;
        q0 += hv * s.e2w[0][u];
        q1 += hv * s.e2w[1][u];
        q2 += hv * s.e2w[2][u];
        q3 += hv * s.e2w[3][u];
      }
      s.u3.part[eb][p8][0] = q0; s.u3.part[eb][p8][1] = q1;
      s.u3.part[eb][p8][2] = q2; s.u3.part[eb][p8][3] = q3;
    }
    __syncthreads();

    if (tid < BT) {   // finalize encoder LSTM2 (H=1)
      float sg[4];
      #pragma unroll
      for (int g = 0; g < 4; ++g) {
        float acc = s.e2b[g] + s.e2whh[g] * s.h2s[tid];
        #pragma unroll
        for (int p = 0; p < 8; ++p) acc += s.u3.part[tid][p][g];
        sg[g] = acc;
      }
      const float cc = fast_sigm(sg[1]) * s.c2s[tid] + fast_sigm(sg[0]) * fast_tanh(sg[2]);
      s.c2s[tid] = cc;
      const float hh = fast_sigm(sg[3]) * fast_tanh(cc);
      s.h2s[tid] = hh;
      s.dec_in[t][tid] = (h16)fmaxf(hh, 0.f);   // relu(enc_out)
    }
    __syncthreads();
  }

  // ---- append x_aux = x[:, -32:, 0] to decoder input ----
  for (int idx = tid; idx < 32 * BT; idx += 128) {
    const int j = idx >> 4, bb = idx & 15;
    s.dec_in[T_ENC + j][bb] = s.u1.xh[bb][64 + j][0];
  }
  for (int i = tid; i < BT * 16; i += 128) (&s.hd[0][0])[i] = (h16)0.0f;

  // decoder LSTM1 B-matrix: aug K = [u(0) | pad(1..15) | h(16..31)]
  V16 bd; float biasd;
  {
    const int gate = wave * 16 + lh;   // wave w = gate type w (i/f/g/o), H=16
    biasd = d1_b[gate];
    #pragma unroll
    for (int i = 0; i < 16; ++i) bd.e[i] = (h16)0.0f;
    if (hi == 0) {
      bd.e[0] = (h16)d1_Wih[gate];     // K0 only
    }
    #pragma unroll
    for (int i = 0; i < 8; ++i) bd.e[8 + i] = (h16)d1_Whh[gate * 16 + hi * 8 + i];  // K16-31
  }
  float cdd[2] = {0.f, 0.f};
  __syncthreads();

  // ================= decoder time loop =================
  for (int t = 0; t < T_DEC; ++t) {
    V16 a;
    a.half8[0] = zero8;
    a.e[0] = (hi == 0) ? s.dec_in[t][lh] : (h16)0.0f;          // K0 = u (lanes<16)
    a.half8[1] = ((const v8h*)&s.hd[lh][0])[hi];               // K16-31 = h0-15
    v8f c;
    #pragma unroll
    for (int i = 0; i < 8; ++i) c[i] = biasd;
    c = __builtin_amdgcn_wmma_f32_16x16x32_f16(false, a.v, false, bd.v, (short)0, c, false, false);
    const int col = wave * 16 + lh;
    #pragma unroll
    for (int r = 0; r < 8; ++r) s.u2.Gd[r + hi * 8][col] = c[r];
    __syncthreads();

    // decoder LSTM1 elementwise fused with decoder LSTM2 partials
    {
      const int bb = tid >> 3;
      float q0 = 0.f, q1 = 0.f, q2 = 0.f, q3 = 0.f;
      #pragma unroll
      for (int q = 0; q < 2; ++q) {
        const int u = (tid & 7) * 2 + q;
        const float gi = s.u2.Gd[bb][u],      gf = s.u2.Gd[bb][16 + u];
        const float gg = s.u2.Gd[bb][32 + u], go = s.u2.Gd[bb][48 + u];
        const float cc = fast_sigm(gf) * cdd[q] + fast_sigm(gi) * fast_tanh(gg);
        cdd[q] = cc;
        const float hv = fast_sigm(go) * fast_tanh(cc);
        s.hd[bb][u] = (h16)hv;
        q0 += hv * s.d2w[0][u];
        q1 += hv * s.d2w[1][u];
        q2 += hv * s.d2w[2][u];
        q3 += hv * s.d2w[3][u];
      }
      s.u3.part[bb][p8][0] = q0; s.u3.part[bb][p8][1] = q1;
      s.u3.part[bb][p8][2] = q2; s.u3.part[bb][p8][3] = q3;
    }
    __syncthreads();

    if (tid < BT) {   // finalize decoder LSTM2 (H=1)
      float sg[4];
      #pragma unroll
      for (int g = 0; g < 4; ++g) {
        float acc = s.d2b[g] + s.d2whh[g] * s.h2d[tid];
        #pragma unroll
        for (int p = 0; p < 8; ++p) acc += s.u3.part[tid][p][g];
        sg[g] = acc;
      }
      const float cc = fast_sigm(sg[1]) * s.c2d[tid] + fast_sigm(sg[0]) * fast_tanh(sg[2]);
      s.c2d[tid] = cc;
      const float hh = fast_sigm(sg[3]) * fast_tanh(cc);
      s.h2d[tid] = hh;
      s.dseq[t][tid] = hh;
    }
    __syncthreads();
  }

  // ================= final FCs =================
  for (int i = tid; i < 32 * 128; i += 128) (&s.u1.fc1w[0][0])[i] = fc1_W[i];  // xh dead now
  for (int i = tid; i < 32 * 32;  i += 128) (&s.u3.fc2.w[0][0])[i] = fc2_W[i]; // part dead now
  if (tid < 32) s.u3.fc2.b[tid] = fc2_b[tid];
  __syncthreads();
  for (int idx = tid; idx < BT * 32; idx += 128) {
    const int bb = idx >> 5, j = idx & 31;
    float acc = fc1_b[j];
    #pragma unroll 8
    for (int k = 0; k < 128; ++k) acc += s.dseq[k][bb] * s.u1.fc1w[j][k];
    s.tmp[bb][j] = acc;
  }
  __syncthreads();
  for (int idx = tid; idx < BT * 32; idx += 128) {
    const int bb = idx >> 5, j = idx & 31;
    float acc = s.u3.fc2.b[j];
    #pragma unroll
    for (int k = 0; k < 32; ++k) acc += s.tmp[bb][k] * s.u3.fc2.w[j][k];
    out[(size_t)(b0 + bb) * 32 + j] = acc;
  }
}

extern "C" void kernel_launch(void* const* d_in, const int* in_sizes, int n_in,
                              void* d_out, int out_size, void* d_ws, size_t ws_size,
                              hipStream_t stream) {
  (void)n_in; (void)out_size; (void)d_ws; (void)ws_size;
  const float* x      = (const float*)d_in[0];
  const float* e1_Wih = (const float*)d_in[1];
  const float* e1_Whh = (const float*)d_in[2];
  const float* e1_b   = (const float*)d_in[3];
  const float* e2_Wih = (const float*)d_in[4];
  const float* e2_Whh = (const float*)d_in[5];
  const float* e2_b   = (const float*)d_in[6];
  const float* d1_Wih = (const float*)d_in[7];
  const float* d1_Whh = (const float*)d_in[8];
  const float* d1_b   = (const float*)d_in[9];
  const float* d2_Wih = (const float*)d_in[10];
  const float* d2_Whh = (const float*)d_in[11];
  const float* d2_b   = (const float*)d_in[12];
  const float* fc1_W  = (const float*)d_in[13];
  const float* fc1_b  = (const float*)d_in[14];
  const float* fc2_W  = (const float*)d_in[15];
  const float* fc2_b  = (const float*)d_in[16];
  float* out = (float*)d_out;

  const int Bsz  = in_sizes[0] / (T_ENC * 8);
  const int grid = Bsz / BT;   // 4096/16 = 256 workgroups
  lstm_encdec_fused<<<grid, 128, 0, stream>>>(
      x, e1_Wih, e1_Whh, e1_b, e2_Wih, e2_Whh, e2_b,
      d1_Wih, d1_Whh, d1_b, d2_Wih, d2_Whh, d2_b,
      fc1_W, fc1_b, fc2_W, fc2_b, out);
}
